// SelfAttention_10187662426545
// MI455X (gfx1250) — matrix operand order repaired and checked by reading, not compiled, not run on hardware
//
#include <hip/hip_runtime.h>
#include <hip/hip_bf16.h>

// ---------------------------------------------------------------------------
// Fused self-attention for MI455X (gfx1250, wave32, WMMA).
//   B=4, S=2048, D=1024, fp32 in/out, bf16 WMMA with f32 accumulation.
// ---------------------------------------------------------------------------

#define BATCH 4
#define SEQ   2048
#define DIM   1024

typedef __attribute__((ext_vector_type(16))) __bf16 v16bf;
typedef __attribute__((ext_vector_type(8)))  __bf16 v8bf;
typedef __attribute__((ext_vector_type(8)))  float  v8f;

__device__ __forceinline__ v8f wmma_bf16(v16bf a, v16bf b, v8f c) {
    // v_wmma_f32_16x16x32_bf16  (D = A(16x32) * B(32x16) + C)
    return __builtin_amdgcn_wmma_f32_16x16x32_bf16(
        /*neg_a=*/false, a, /*neg_b=*/false, b,
        /*c_mod=*/(short)0, c, /*reuse_a=*/false, /*reuse_b=*/false);
}

// A-matrix fragment (16x32, bf16), per CDNA5 layout:
//   lane L: M = L&15, kgrp = (L>>4)*8
//   halves 0..7  -> K = kbase + kgrp + 0..7
//   halves 8..15 -> K = kbase + 16 + kgrp + 0..7
__device__ __forceinline__ v16bf load_a_frag(const __bf16* base, int lane,
                                             int ldk, int kbase) {
    const int M  = lane & 15;
    const int kg = (lane >> 4) * 8;
    const __bf16* p = base + (size_t)M * ldk + kbase + kg;
    union { v16bf v; v8bf h[2]; } u;
    u.h[0] = *(const v8bf*)(p);
    u.h[1] = *(const v8bf*)(p + 16);
    return u.v;
}

// B-matrix fragment (32x16, bf16) sourced from a row-major [N, K] matrix
// (i.e. the transposed operand: B(k,n) = Src[n,k]).  Per CDNA5 layout:
//   lane L: N = n0 + (L&15), kb = (L>>4)*8, halves 0..7 -> K = kbase+kb+0..7,
//   halves 8..15 -> K = kbase+16+kb+0..7 (matches A fragment order).
__device__ __forceinline__ v16bf load_b_frag(const __bf16* base, int lane,
                                             int ldk, int n0, int kbase) {
    const int n  = n0 + (lane & 15);
    const int kb = (lane >> 4) * 8;
    const __bf16* p = base + (size_t)n * ldk + kbase + kb;
    union { v16bf v; v8bf h[2]; } u;
    u.h[0] = *(const v8bf*)(p);
    u.h[1] = *(const v8bf*)(p + 16);
    return u.v;
}

// ---------------------------------------------------------------------------
// Kernel 0: fp32 -> bf16 conversion (x and the three weight matrices)
// ---------------------------------------------------------------------------
__global__ void cvt_f32_bf16(const float* __restrict__ src,
                             __bf16* __restrict__ dst, int n) {
    int i = blockIdx.x * blockDim.x + threadIdx.x;
    if (i < n) dst[i] = (__bf16)src[i];
}

// ---------------------------------------------------------------------------
// Kernel 1: QKV projection.  grid = (B*S/16, 3), block = 256 (8 waves).
//   Each workgroup: one 16-row strip x all 1024 columns for one projection.
//   Wave w owns columns [w*128, w*128+128) -> 8 accumulator tiles.
//   No LDS, ~90 VGPRs -> high occupancy; TLP hides load latency here, so no
//   software pipeline (keeps register count low).
//   proj 0 -> Q (row-major bf16), 1 -> K (row-major), 2 -> V (Vt[B,D,S]).
// ---------------------------------------------------------------------------
__global__ void __launch_bounds__(256)
qkv_proj(const __bf16* __restrict__ xb,
         const __bf16* __restrict__ wq,
         const __bf16* __restrict__ wk,
         const __bf16* __restrict__ wv,
         const float* __restrict__ bq,
         const float* __restrict__ bk,
         const float* __restrict__ bv,
         __bf16* __restrict__ Qb,
         __bf16* __restrict__ Kb,
         __bf16* __restrict__ Vt) {
    const int m0   = blockIdx.x * 16;            // global row block
    const int proj = blockIdx.y;                 // 0=Q 1=K 2=V
    const int wave = threadIdx.x >> 5;
    const int lane = threadIdx.x & 31;
    const int n0   = wave * 128;

    const __bf16* W    = (proj == 0) ? wq : (proj == 1) ? wk : wv;
    const float*  bias = (proj == 0) ? bq : (proj == 1) ? bk : bv;
    const __bf16* Arow = xb + (size_t)m0 * DIM;

    const v8f zero = {0.f, 0.f, 0.f, 0.f, 0.f, 0.f, 0.f, 0.f};
    v8f acc[8];
#pragma unroll
    for (int nt = 0; nt < 8; ++nt) acc[nt] = zero;

    for (int dc = 0; dc < DIM / 32; ++dc) {
        v16bf a = load_a_frag(Arow, lane, DIM, dc * 32);
#pragma unroll
        for (int nt = 0; nt < 8; ++nt) {
            v16bf b = load_b_frag(W, lane, DIM, n0 + nt * 16, dc * 32);
            acc[nt] = wmma_bf16(a, b, acc[nt]);
        }
    }

    // C/D layout: lane L, vgpr r -> M = r + 8*(L>>4), N = L&15
    const int N    = lane & 15;
    const int rOff = (lane >> 4) * 8;
#pragma unroll
    for (int nt = 0; nt < 8; ++nt) {
        const int  ncol = n0 + nt * 16 + N;
        const float bv_ = bias[ncol];
        if (proj < 2) {
            __bf16* dst = (proj == 0) ? Qb : Kb;
#pragma unroll
            for (int r = 0; r < 8; ++r)
                dst[(size_t)(m0 + rOff + r) * DIM + ncol] =
                    (__bf16)(acc[nt][r] + bv_);
        } else {
            const int b  = m0 >> 11;                 // m0 / SEQ
            const int s0 = (m0 & (SEQ - 1)) + rOff;  // row within batch
            union { v8bf v; __bf16 e[8]; } pk;
#pragma unroll
            for (int r = 0; r < 8; ++r) pk.e[r] = (__bf16)(acc[nt][r] + bv_);
            *(v8bf*)&Vt[(size_t)b * DIM * SEQ + (size_t)ncol * SEQ + s0] = pk.v;
        }
    }
}

// ---------------------------------------------------------------------------
// Kernel 2: attention.  grid = B*S/16 (one 16-query block per workgroup),
// block = 256 (8 waves).  Dynamic LDS: 128KB f32 scores + 64KB bf16 P
// -> 1 WG/WGP (2 waves/SIMD), so both GEMM phases use an explicit 2-stage
// software pipeline (ping-pong fragment buffers with overlapping lifetimes)
// to force partial s_wait_loadcnt and overlap VMEM with the matrix pipe.
// ---------------------------------------------------------------------------
__global__ void __launch_bounds__(256)
attention(const __bf16* __restrict__ Qb,
          const __bf16* __restrict__ Kb,
          const __bf16* __restrict__ Vt,
          float* __restrict__ out) {
    extern __shared__ char smem[];
    float*  sS = (float*)smem;                       // [16][SEQ] f32  (128KB)
    __bf16* sP = (__bf16*)(smem + 16 * SEQ * 4);     // [16][SEQ] bf16 ( 64KB)

    const int b    = blockIdx.x >> 7;                // SEQ/16 = 128 blocks/batch
    const int q0   = (blockIdx.x & 127) * 16;
    const int wave = threadIdx.x >> 5;
    const int lane = threadIdx.x & 31;

    const __bf16* Qbase  = Qb + ((size_t)b * SEQ + q0) * DIM;
    const __bf16* Kbase  = Kb + (size_t)b * SEQ * DIM;
    const __bf16* Vtbase = Vt + (size_t)b * DIM * SEQ;

    const v8f zero = {0.f, 0.f, 0.f, 0.f, 0.f, 0.f, 0.f, 0.f};
    const int N    = lane & 15;
    const int rOff = (lane >> 4) * 8;

    // ---- Phase 1: scores = Q @ K^T (raw dot; 1/sqrt(D) folded into softmax)
    //      Wave w covers keys [w*256, w*256+256): 4 groups of 4 key tiles,
    //      software-pipelined over the 32 K-chunks.
    for (int tg = 0; tg < 4; ++tg) {
        const int key0 = wave * 256 + tg * 64;
        v8f acc[4];
#pragma unroll
        for (int j = 0; j < 4; ++j) acc[j] = zero;

        v16bf aA = load_a_frag(Qbase, lane, DIM, 0);
        v16bf kA[4];
#pragma unroll
        for (int j = 0; j < 4; ++j)
            kA[j] = load_b_frag(Kbase, lane, DIM, key0 + j * 16, 0);

        for (int dc = 0; dc < DIM / 32; dc += 2) {
            const int k1 = (dc + 1) * 32;
            const int k2 = ((dc + 2 < DIM / 32) ? (dc + 2) : (DIM / 32 - 1)) * 32;

            v16bf aB = load_a_frag(Qbase, lane, DIM, k1);
            v16bf kB[4];
#pragma unroll
            for (int j = 0; j < 4; ++j)
                kB[j] = load_b_frag(Kbase, lane, DIM, key0 + j * 16, k1);
#pragma unroll
            for (int j = 0; j < 4; ++j)
                acc[j] = wmma_bf16(aA, kA[j], acc[j]);

            aA = load_a_frag(Qbase, lane, DIM, k2);
#pragma unroll
            for (int j = 0; j < 4; ++j)
                kA[j] = load_b_frag(Kbase, lane, DIM, key0 + j * 16, k2);
#pragma unroll
            for (int j = 0; j < 4; ++j)
                acc[j] = wmma_bf16(aB, kB[j], acc[j]);
        }
#pragma unroll
        for (int j = 0; j < 4; ++j)
#pragma unroll
            for (int r = 0; r < 8; ++r)
                sS[(rOff + r) * SEQ + key0 + j * 16 + N] = acc[j][r];
    }
    __syncthreads();

    // ---- Phase 2: safe softmax, 16 threads per row
    {
        const float scale = 0.03125f;                // 1/sqrt(1024)
        const int   row   = threadIdx.x >> 4;
        const int   sub   = threadIdx.x & 15;
        const float* rp   = sS + row * SEQ;

        float m = -3.0e38f;
        for (int c = sub; c < SEQ; c += 16) m = fmaxf(m, rp[c]);
#pragma unroll
        for (int msk = 1; msk < 16; msk <<= 1)
            m = fmaxf(m, __shfl_xor(m, msk, 16));

        float s = 0.f;
        for (int c = sub; c < SEQ; c += 16) s += __expf((rp[c] - m) * scale);
#pragma unroll
        for (int msk = 1; msk < 16; msk <<= 1)
            s += __shfl_xor(s, msk, 16);

        const float inv = 1.f / s;
        for (int c = sub; c < SEQ; c += 16)
            sP[row * SEQ + c] = (__bf16)(__expf((rp[c] - m) * scale) * inv);
    }
    __syncthreads();

    // ---- Phase 3: O = P @ V, wave owns 128 output columns,
    //      software-pipelined over the 64 key steps of 32.
    const int d0 = wave * 128;
    v8f acc[8];
#pragma unroll
    for (int nt = 0; nt < 8; ++nt) acc[nt] = zero;

    v16bf pA = load_a_frag(sP, lane, SEQ, 0);        // ds_load_b128 x2
    v16bf vA[8];
#pragma unroll
    for (int nt = 0; nt < 8; ++nt)
        vA[nt] = load_b_frag(Vtbase, lane, SEQ, d0 + nt * 16, 0);

    for (int ks = 0; ks < SEQ / 32; ks += 2) {
        const int k1 = (ks + 1) * 32;
        const int k2 = ((ks + 2 < SEQ / 32) ? (ks + 2) : (SEQ / 32 - 1)) * 32;

        v16bf pB = load_a_frag(sP, lane, SEQ, k1);
        v16bf vB[8];
#pragma unroll
        for (int nt = 0; nt < 8; ++nt)
            vB[nt] = load_b_frag(Vtbase, lane, SEQ, d0 + nt * 16, k1);
#pragma unroll
        for (int nt = 0; nt < 8; ++nt)
            acc[nt] = wmma_bf16(pA, vA[nt], acc[nt]);

        pA = load_a_frag(sP, lane, SEQ, k2);
#pragma unroll
        for (int nt = 0; nt < 8; ++nt)
            vA[nt] = load_b_frag(Vtbase, lane, SEQ, d0 + nt * 16, k2);
#pragma unroll
        for (int nt = 0; nt < 8; ++nt)
            acc[nt] = wmma_bf16(pB, vB[nt], acc[nt]);
    }

#pragma unroll
    for (int nt = 0; nt < 8; ++nt) {
        const int dcol = d0 + nt * 16 + N;
#pragma unroll
        for (int r = 0; r < 8; ++r)
            out[((size_t)(b * SEQ + q0 + rOff + r)) * DIM + dcol] = acc[nt][r];
    }
}

// ---------------------------------------------------------------------------
// Host entry
// ---------------------------------------------------------------------------
extern "C" void kernel_launch(void* const* d_in, const int* in_sizes, int n_in,
                              void* d_out, int out_size, void* d_ws, size_t ws_size,
                              hipStream_t stream) {
    const float* x  = (const float*)d_in[0];
    const float* Wq = (const float*)d_in[1];
    const float* bq = (const float*)d_in[2];
    const float* Wk = (const float*)d_in[3];
    const float* bk = (const float*)d_in[4];
    const float* Wv = (const float*)d_in[5];
    const float* bv = (const float*)d_in[6];
    float* out = (float*)d_out;

    const size_t MB = (size_t)1 << 20;
    char* ws = (char*)d_ws;
    __bf16* Qb  = (__bf16*)(ws + 0 * MB);    // [B*S, D]   16 MB
    __bf16* Kb  = (__bf16*)(ws + 16 * MB);   // [B*S, D]   16 MB
    __bf16* Vt  = (__bf16*)(ws + 32 * MB);   // [B, D, S]  16 MB
    __bf16* xb  = (__bf16*)(ws + 48 * MB);   // [B*S, D]   16 MB
    __bf16* wqb = (__bf16*)(ws + 64 * MB);   // [D, D]      2 MB
    __bf16* wkb = (__bf16*)(ws + 66 * MB);
    __bf16* wvb = (__bf16*)(ws + 68 * MB);

    const int nx = BATCH * SEQ * DIM;        // 8388608
    const int nw = DIM * DIM;                // 1048576

    cvt_f32_bf16<<<(nx + 255) / 256, 256, 0, stream>>>(x,  xb,  nx);
    cvt_f32_bf16<<<(nw + 255) / 256, 256, 0, stream>>>(Wq, wqb, nw);
    cvt_f32_bf16<<<(nw + 255) / 256, 256, 0, stream>>>(Wk, wkb, nw);
    cvt_f32_bf16<<<(nw + 255) / 256, 256, 0, stream>>>(Wv, wvb, nw);

    dim3 g1(BATCH * SEQ / 16, 3);            // 512 x 3 workgroups
    qkv_proj<<<g1, 256, 0, stream>>>(xb, wqb, wkb, wvb, bq, bk, bv, Qb, Kb, Vt);

    const size_t ldsBytes = (size_t)16 * SEQ * 4 + (size_t)16 * SEQ * 2; // 192KB
    attention<<<BATCH * SEQ / 16, 256, ldsBytes, stream>>>(Qb, Kb, Vt, out);

    (void)in_sizes; (void)n_in; (void)out_size; (void)ws_size;
}
